// GNNSequenceProcessor_60473139528095
// MI455X (gfx1250) — matrix operand order repaired and checked by reading, not compiled, not run on hardware
//
#include <hip/hip_runtime.h>

// ---------------------------------------------------------------------------
// GCN 2-layer chain:  t = h@W ;  agg = a_i*t[i] + c_i*t[i-1] ; h = relu(agg+b)
// bf16x3-split WMMA GEMM (v_wmma_f32_16x16x32_bf16) + tiny chain fix kernel.
// A is split fp32->bf16(hi,lo) ONCE at LDS-staging time; consumers build
// fragments with pure ds_load_b128 (no VALU). B planes are pre-transposed /
// pre-split to [N][K] bf16 in ws and fetched straight from global (L2-resident).
// ---------------------------------------------------------------------------

typedef __attribute__((ext_vector_type(16))) __bf16 v16bf;
typedef __attribute__((ext_vector_type(8)))  float  v8f;

#define NROWS   65536
#define DOUT    512
#define CHAIN   4096      // rows touched by the chain graph
#define BM      64        // rows per block
#define BK      32        // K per WMMA step
#define LDAB    40        // padded LDS row stride (bf16 elems): 80B, spreads banks

union Frag { v16bf v; uint4 q[2]; __bf16 e[16]; };

// split 8 floats into bf16 hi/lo planes and store 16B each
__device__ __forceinline__ void split_store(float4 a, float4 b,
                                            __bf16* __restrict__ hi,
                                            __bf16* __restrict__ lo)
{
    union { __bf16 e[8]; uint4 q; } H, L;
    const float f[8] = { a.x, a.y, a.z, a.w, b.x, b.y, b.z, b.w };
#pragma unroll
    for (int e = 0; e < 8; ++e) {
        __bf16 h = (__bf16)f[e];
        H.e[e] = h;
        L.e[e] = (__bf16)(f[e] - (float)h);
    }
    *(uint4*)hi = H.q;
    *(uint4*)lo = L.q;
}

// ---------------------------------------------------------------------------
// Weight prep: W[K][N] fp32  ->  WtHi/WtLo [N][K] bf16 (hi/lo split, RNE)
// ---------------------------------------------------------------------------
__global__ void prep_w(const float* __restrict__ W,
                       __bf16* __restrict__ Hi, __bf16* __restrict__ Lo,
                       int K, int Nc)
{
    int idx = blockIdx.x * blockDim.x + threadIdx.x;
    if (idx >= K * Nc) return;
    int k = idx / Nc;
    int n = idx - k * Nc;
    float  w = W[idx];
    __bf16 h = (__bf16)w;
    __bf16 l = (__bf16)(w - (float)h);
    Hi[(size_t)n * K + k] = h;
    Lo[(size_t)n * K + k] = l;
}

// ---------------------------------------------------------------------------
// Fused GEMM:  t = A @ W  via bf16x3 WMMA.
//   rows >= CHAIN : out = relu(t + bias)          (deg==1, no neighbors)
//   rows <  CHAIN : traw = t                      (fixed up by gcn_fix)
// ---------------------------------------------------------------------------
template<int KD>
__global__ __launch_bounds__(256)
void gcn_gemm(const float*  __restrict__ A,     // [NROWS, KD]
              const __bf16* __restrict__ WtHi,  // [DOUT, KD]
              const __bf16* __restrict__ WtLo,  // [DOUT, KD]
              const float*  __restrict__ bias,  // [DOUT]
              float* __restrict__ out,          // [NROWS, DOUT]
              float* __restrict__ traw)         // [CHAIN, DOUT]
{
    __shared__ __bf16 AsHi[2][BM][LDAB];
    __shared__ __bf16 AsLo[2][BM][LDAB];

    const int tid  = threadIdx.x;
    const int lane = tid & 31;
    const int wave = tid >> 5;
    const int r0   = blockIdx.x * BM;

    const int wm   = wave >> 2;          // 0..1  (32 rows each)
    const int wn   = wave & 3;           // 0..3  (128 cols each)
    const int hsel = lane >> 4;          // high half of wave
    const int mrow = lane & 15;

    // global->LDS A staging: thread t handles 8 floats of row t/4
    const int arow = tid >> 2;
    const int aseg = (tid & 3) * 8;
    const float* aptr = A + (size_t)(r0 + arow) * KD + aseg;

    constexpr int NSTEP = KD / BK;

    // prologue: stage k-step 0 (split once here, not per consumer)
    float4 pf0 = *(const float4*)(aptr + 0);
    float4 pf1 = *(const float4*)(aptr + 4);
    split_store(pf0, pf1, &AsHi[0][arow][aseg], &AsLo[0][arow][aseg]);
    __syncthreads();

    v8f acc[2][8];
#pragma unroll
    for (int m = 0; m < 2; ++m)
#pragma unroll
        for (int n = 0; n < 8; ++n)
#pragma unroll
            for (int g = 0; g < 8; ++g) acc[m][n][g] = 0.0f;

    for (int ks = 0; ks < NSTEP; ++ks) {
        const int buf = ks & 1;

        // prefetch next A tile into registers while we compute
        if (ks + 1 < NSTEP) {
            const float* p = aptr + (size_t)(ks + 1) * BK;
            pf0 = *(const float4*)(p + 0);
            pf1 = *(const float4*)(p + 4);
        }

        // ---- A fragments: pure b128 LDS loads, contiguous k order (same as B)
        // lane l: row = l&15; elems 0..15 = K(16h..16h+15)
        Frag ah[2], al[2];
#pragma unroll
        for (int m = 0; m < 2; ++m) {
            const int row = wm * 32 + m * 16 + mrow;
            ah[m].q[0] = *(const uint4*)&AsHi[buf][row][16 * hsel];
            ah[m].q[1] = *(const uint4*)&AsHi[buf][row][16 * hsel + 8];
            al[m].q[0] = *(const uint4*)&AsLo[buf][row][16 * hsel];
            al[m].q[1] = *(const uint4*)&AsLo[buf][row][16 * hsel + 8];
        }

        // ---- B fragments straight from global (pre-transposed Wt planes):
        // lane n<16 holds col n K=kb..kb+15; lane n+16 holds K=kb+16..kb+31
        const int koff = ks * BK + 16 * hsel;
#pragma unroll
        for (int n = 0; n < 8; ++n) {
            const int col = wn * 128 + n * 16 + (lane & 15);
            const __bf16* bh = WtHi + (size_t)col * KD + koff;
            const __bf16* bl = WtLo + (size_t)col * KD + koff;
            Frag fbh, fbl;
            fbh.q[0] = ((const uint4*)bh)[0];
            fbh.q[1] = ((const uint4*)bh)[1];
            fbl.q[0] = ((const uint4*)bl)[0];
            fbl.q[1] = ((const uint4*)bl)[1];
#pragma unroll
            for (int m = 0; m < 2; ++m) {
                acc[m][n] = __builtin_amdgcn_wmma_f32_16x16x32_bf16(
                    false, ah[m].v, false, fbh.v, (short)0, acc[m][n], false, false);
                acc[m][n] = __builtin_amdgcn_wmma_f32_16x16x32_bf16(
                    false, ah[m].v, false, fbl.v, (short)0, acc[m][n], false, false);
                acc[m][n] = __builtin_amdgcn_wmma_f32_16x16x32_bf16(
                    false, al[m].v, false, fbh.v, (short)0, acc[m][n], false, false);
            }
        }

        // write prefetched tile to the other buffer, then one barrier/step
        if (ks + 1 < NSTEP) {
            split_store(pf0, pf1,
                        &AsHi[buf ^ 1][arow][aseg], &AsLo[buf ^ 1][arow][aseg]);
        }
        __syncthreads();
    }

    // ---- epilogue: C layout = VGPR g -> row m0+g+8*hsel, col = n0+(lane&15)
    const bool low = (r0 < CHAIN);   // block-uniform
#pragma unroll
    for (int m = 0; m < 2; ++m) {
        const int R0 = r0 + wm * 32 + m * 16 + hsel * 8;
#pragma unroll
        for (int n = 0; n < 8; ++n) {
            const int col = wn * 128 + n * 16 + (lane & 15);
            const float bv = bias[col];
#pragma unroll
            for (int g = 0; g < 8; ++g) {
                const size_t row = (size_t)(R0 + g);
                const float t = acc[m][n][g];
                if (low) {
                    traw[row * DOUT + col] = t;       // raw t, fixed later
                } else {
                    const float v = t + bv;           // deg==1: agg == t
                    out[row * DOUT + col] = v > 0.0f ? v : 0.0f;
                }
            }
        }
    }
}

// ---------------------------------------------------------------------------
// Chain aggregation for rows 0..CHAIN-1:
//   i==0      : agg = t[0]
//   i==1      : agg = 0.5*t[1] + rsqrt(2)*t[0]
//   2..4095   : agg = 0.5*t[i] + 0.5*t[i-1]
// then relu(+bias).
// ---------------------------------------------------------------------------
__global__ __launch_bounds__(256)
void gcn_fix(const float* __restrict__ traw, const float* __restrict__ bias,
             float* __restrict__ out)
{
    const int idx = blockIdx.x * blockDim.x + threadIdx.x;  // CHAIN*DOUT
    const int i = idx >> 9;
    const int c = idx & (DOUT - 1);
    const float t    = traw[idx];
    const float prev = (i == 0) ? 0.0f : traw[idx - DOUT];
    const float a    = (i == 0) ? 1.0f : 0.5f;
    const float cc   = (i == 0) ? 0.0f
                     : (i == 1 ? 0.70710678118654752440f : 0.5f);
    const float v = a * t + cc * prev + bias[c];
    out[idx] = v > 0.0f ? v : 0.0f;
}

// ---------------------------------------------------------------------------
extern "C" void kernel_launch(void* const* d_in, const int* in_sizes, int n_in,
                              void* d_out, int out_size, void* d_ws, size_t ws_size,
                              hipStream_t stream)
{
    const float* x  = (const float*)d_in[0];   // [16,4096,256]
    const float* W1 = (const float*)d_in[1];   // [256,512]
    const float* b1 = (const float*)d_in[2];   // [512]
    const float* W2 = (const float*)d_in[3];   // [512,512]
    const float* b2 = (const float*)d_in[4];   // [512]
    float* out = (float*)d_out;                // [16,4096,512]

    // ws layout (bytes): Wt1Hi | Wt1Lo | Wt2Hi | Wt2Lo | traw  (~10 MB total)
    char* ws = (char*)d_ws;
    __bf16* Wt1Hi = (__bf16*)(ws + 0);                       // 512*256*2
    __bf16* Wt1Lo = (__bf16*)(ws + 262144);
    __bf16* Wt2Hi = (__bf16*)(ws + 524288);                  // 512*512*2
    __bf16* Wt2Lo = (__bf16*)(ws + 1048576);
    float*  traw  = (float*) (ws + 1572864);                 // 4096*512*4

    prep_w<<<(256 * 512 + 255) / 256, 256, 0, stream>>>(W1, Wt1Hi, Wt1Lo, 256, 512);
    prep_w<<<(512 * 512 + 255) / 256, 256, 0, stream>>>(W2, Wt2Hi, Wt2Lo, 512, 512);

    // Layer 1: x -> h1 (stored in d_out; fully overwritten before layer 2 reads)
    gcn_gemm<256><<<NROWS / BM, 256, 0, stream>>>(x, Wt1Hi, Wt1Lo, b1, out, traw);
    gcn_fix<<<(CHAIN * DOUT) / 256, 256, 0, stream>>>(traw, b1, out);

    // Layer 2: in-place on d_out (each block reads only the rows it writes)
    gcn_gemm<512><<<NROWS / BM, 256, 0, stream>>>(out, Wt2Hi, Wt2Lo, b2, out, traw);
    gcn_fix<<<(CHAIN * DOUT) / 256, 256, 0, stream>>>(traw, b2, out);
}
